// fn_blstm_76278619177029
// MI455X (gfx1250) — matrix operand order repaired and checked by reading, not compiled, not run on hardware
//
#include <hip/hip_runtime.h>

#define Vv  64
#define Hh  128
#define Bb  128
#define Tt  512
#define G4H 512

typedef __attribute__((ext_vector_type(16))) __bf16 v16bf;
typedef __attribute__((ext_vector_type(8)))  __bf16 v8bf;
typedef __attribute__((ext_vector_type(8)))  float  v8f;

__device__ __forceinline__ float fast_tanh(float x) {
#if __has_builtin(__builtin_amdgcn_tanh_f32)
  return __builtin_amdgcn_tanh_f32(x);     // native V_TANH_F32 on gfx1250
#elif __has_builtin(__builtin_amdgcn_tanhf)
  return __builtin_amdgcn_tanhf(x);
#else
  float e = __expf(-2.0f * x);
  return (1.0f - e) / (1.0f + e);
#endif
}
__device__ __forceinline__ float fast_sigmoid(float x) {
  return 0.5f * fast_tanh(0.5f * x) + 0.5f;
}

// Assemble the 16-bit A-operand for one lane from its two contiguous 8-element
// K-runs (ISA 16-bit A 16x32 layout: VGPR0-3 = K run at hi*8, VGPR4-7 = +16).
__device__ __forceinline__ v16bf make_a16(const __bf16* run1, const __bf16* run2) {
  v8bf a0 = *(const v8bf*)run1;
  v8bf a1 = *(const v8bf*)run2;
  v16bf a;
#pragma unroll
  for (int i = 0; i < 8; ++i) { a[i] = a0[i]; a[8 + i] = a1[i]; }
  return a;
}

// ---------------------------------------------------------------------------
// Prep: f32 -> bf16 weight/emb conversion + combined bias
// ---------------------------------------------------------------------------
__global__ __launch_bounds__(256) void prep_kernel(
    const float* __restrict__ emb,
    const float* __restrict__ wihf, const float* __restrict__ whhf,
    const float* __restrict__ bihf, const float* __restrict__ bhhf,
    const float* __restrict__ wihb, const float* __restrict__ whhb,
    const float* __restrict__ bihb, const float* __restrict__ bhhb,
    const float* __restrict__ fcw,
    __bf16* __restrict__ emb_bf, __bf16* __restrict__ wih_bf,
    __bf16* __restrict__ whh_bf, __bf16* __restrict__ fcw_bf,
    float* __restrict__ bsum) {
  int id = blockIdx.x * blockDim.x + threadIdx.x;  // 0..65535
  if (id < Vv * Hh) emb_bf[id] = (__bf16)emb[id];
  if (id < G4H * Hh) {
    wih_bf[id] = (__bf16)wihf[id];
    wih_bf[G4H * Hh + id] = (__bf16)wihb[id];
    whh_bf[id] = (__bf16)whhf[id];
    whh_bf[G4H * Hh + id] = (__bf16)whhb[id];
  }
  if (id < Vv * 2 * Hh) fcw_bf[id] = (__bf16)fcw[id];
  if (id < G4H) {
    bsum[id] = bihf[id] + bhhf[id];
    bsum[G4H + id] = bihb[id] + bhhb[id];
  }
}

// ---------------------------------------------------------------------------
// Phase 1: xp = emb[x] @ w_ih^T + (b_ih+b_hh), stored SWIZZLED in C-matrix
// lane layout: xp[t][pair p=mtl*8+ht][gate g][lane*8 + v]  (bf16)
// grid (T*B/16, 512/(16*8), 2dir), block 256 (8 waves, one 16x16 tile each)
// ---------------------------------------------------------------------------
__global__ __launch_bounds__(256) void proj_kernel(
    const int* __restrict__ x, const __bf16* __restrict__ emb_bf,
    const __bf16* __restrict__ wih_bf, const float* __restrict__ bsum,
    __bf16* __restrict__ xp) {
  const int lane = threadIdx.x & 31;
  const int w = threadIdx.x >> 5;
  const int lo = lane & 15, hi = lane >> 4;
  const int dir = blockIdx.z;
  const int row0 = blockIdx.x * 16;          // global m-tile = t*8 + mtl
  const int ntile = blockIdx.y * 8 + w;      // 0..31
  const int n0 = ntile * 16;

  const __bf16* wihd = wih_bf + (size_t)dir * G4H * Hh;
  const float* bs = bsum + dir * G4H;
  __bf16* xpd = xp + (size_t)dir * Tt * Bb * G4H;

  // A-row for this lane: row r = t*B + b
  const int r = row0 + lo;
  const int t = r >> 7, b = r & 127;
  const __bf16* erow = emb_bf + x[b * Tt + t] * Hh;
  const __bf16* brow = wihd + (size_t)(n0 + lo) * Hh;  // B column = w_ih row

  const float bias = bs[n0 + lo];
  v8f acc;
#pragma unroll
  for (int v = 0; v < 8; ++v) acc[v] = bias;

#pragma unroll
  for (int kb = 0; kb < 4; ++kb) {
    const int k0 = kb * 32;
    v16bf a = make_a16(erow + k0 + hi * 8, erow + k0 + 16 + hi * 8);
    v16bf bm = make_a16(brow + k0 + hi * 8, brow + k0 + 16 + hi * 8);
    acc = __builtin_amdgcn_wmma_f32_16x16x32_bf16(false, a, false, bm,
                                                  (short)0, acc, false, false);
  }

  // swizzled store: one contiguous 16B b128 store per lane
  const int mtl = blockIdx.x & 7;            // m-tile within this t
  const int g = ntile >> 3, ht = ntile & 7;  // gate, h-tile
  const int p = mtl * 8 + ht;                // pair index 0..63
  v8bf outv;
#pragma unroll
  for (int v = 0; v < 8; ++v) outv[v] = (__bf16)acc[v];
  const int tt = row0 >> 7;                  // == t (uniform over tile)
  *(v8bf*)(xpd + ((((size_t)tt * 64 + p) * 4 + g) * 256) + lane * 8) = outv;
}

// ---------------------------------------------------------------------------
// Phase 2: persistent recurrent kernel. One workgroup (1024 thr = 32 waves)
// per direction. w_hh in LDS (128KB), h double-buffered in LDS (64KB),
// c in registers. Each wave owns 2 (m-tile,h-tile) pairs; for each pair it
// computes all four gate tiles so i,f,g,o land in the same lane. xp is read
// in the swizzled C layout (contiguous per lane) and added AFTER the WMMA
// chain so the global loads hide under the LDS-fed WMMAs.
// ---------------------------------------------------------------------------
__global__ __launch_bounds__(1024) void lstm_kernel(
    const __bf16* __restrict__ xp_all, const __bf16* __restrict__ whh_bf,
    __bf16* __restrict__ hs_all) {
  extern __shared__ __align__(16) char smem[];
  __bf16* whh = (__bf16*)smem;                         // 512*128 bf16 = 128KB
  __bf16* hbuf = (__bf16*)(smem + G4H * Hh * 2);       // 2*128*128 bf16 = 64KB

  const int dir = blockIdx.x;
  const int tid = threadIdx.x;
  const int lane = tid & 31, w = tid >> 5;
  const int lo = lane & 15, hi = lane >> 4;

  const __bf16* xp = xp_all + (size_t)dir * Tt * Bb * G4H;
  const __bf16* whhg = whh_bf + (size_t)dir * G4H * Hh;
  __bf16* hs = hs_all + (size_t)dir * Tt * Bb * Hh;

  {  // stage w_hh into LDS; zero h buffer 0
    const uint4* src = (const uint4*)whhg;
    uint4* dst = (uint4*)whh;
#pragma unroll
    for (int i = 0; i < 8; ++i) dst[tid + i * 1024] = src[tid + i * 1024];
    unsigned int* hz = (unsigned int*)hbuf;
#pragma unroll
    for (int i = 0; i < 8; ++i) hz[tid + i * 1024] = 0u;
  }
  __syncthreads();

  float creg[2][8];
#pragma unroll
  for (int p = 0; p < 2; ++p)
#pragma unroll
    for (int v = 0; v < 8; ++v) creg[p][v] = 0.f;

  for (int s = 0; s < Tt; ++s) {
    const int t = dir ? (Tt - 1 - s) : s;
    const __bf16* xpt = xp + (size_t)t * Bb * G4H;     // swizzled slab, 128KB
    const __bf16* rbuf = hbuf + (s & 1) * (Bb * Hh);
    __bf16* wbuf = hbuf + ((s + 1) & 1) * (Bb * Hh);

    // pull next timestep's gate-input slab toward the caches
    const int tn = dir ? (t - 1) : (t + 1);
    if (tn >= 0 && tn < Tt)
      __builtin_prefetch(xp + (size_t)tn * Bb * G4H + tid * 64, 0, 1);

#pragma unroll
    for (int pp = 0; pp < 2; ++pp) {
      const int p = w + pp * 32;          // 0..63 (m-tile, h-tile) pair index
      const int mt = p >> 3, ht = p & 7;
      const int mbase = mt * 16;
      const int hh0 = ht * 16 + lo;       // this lane's h column

      // issue xp loads early (contiguous 16B per gate), independent of WMMAs
      v8bf xv[4];
#pragma unroll
      for (int g = 0; g < 4; ++g)
        xv[g] = *(const v8bf*)(xpt + (((size_t)p * 4 + g) * 256) + lane * 8);

      v8f acc[4];
#pragma unroll
      for (int g = 0; g < 4; ++g)
#pragma unroll
        for (int v = 0; v < 8; ++v) acc[g][v] = 0.f;

      const __bf16* arow = rbuf + (mbase + lo) * Hh;  // h_{t-1} row for A
#pragma unroll
      for (int kb = 0; kb < 4; ++kb) {
        const int k0 = kb * 32;
        v16bf a = make_a16(arow + k0 + hi * 8, arow + k0 + 16 + hi * 8);
#pragma unroll
        for (int g = 0; g < 4; ++g) {
          const __bf16* brow = whh + (size_t)(g * Hh + hh0) * Hh + k0 + hi * 8;
          v16bf bm = make_a16(brow, brow + 16);
          acc[g] = __builtin_amdgcn_wmma_f32_16x16x32_bf16(
              false, a, false, bm, (short)0, acc[g], false, false);
        }
      }

      // cell update: lane holds i,f,g,o for the same (b,h) -> no shuffles
#pragma unroll
      for (int v = 0; v < 8; ++v) {
        const float ig = fast_sigmoid(acc[0][v] + (float)xv[0][v]);
        const float fg = fast_sigmoid(acc[1][v] + (float)xv[1][v]);
        const float gg = fast_tanh(acc[2][v] + (float)xv[2][v]);
        const float og = fast_sigmoid(acc[3][v] + (float)xv[3][v]);
        const float c = fg * creg[pp][v] + ig * gg;
        creg[pp][v] = c;
        const float h = og * fast_tanh(c);
        const int m = mbase + hi * 8 + v;
        const __bf16 hb = (__bf16)h;
        wbuf[m * Hh + hh0] = hb;
        hs[(size_t)t * Bb * Hh + m * Hh + hh0] = hb;
      }
    }
    __syncthreads();  // single barrier/step thanks to double-buffered h
  }
}

// ---------------------------------------------------------------------------
// Phase 3: out[b,t,:] = [h_f ; h_b] @ fc_w^T + fc_b   (bf16 WMMA)
// Row ordering r = b*T + t so each m-tile's outputs (4 waves) cover one contiguous 4KB.
// grid 2048, block 256: 8 waves cover 2 m-tiles x 4 n-tiles
// ---------------------------------------------------------------------------
__global__ __launch_bounds__(256) void fc_kernel(
    const __bf16* __restrict__ hs_all, const __bf16* __restrict__ fcw_bf,
    const float* __restrict__ fcb, float* __restrict__ out) {
  const int lane = threadIdx.x & 31, w = threadIdx.x >> 5;
  const int lo = lane & 15, hi = lane >> 4;
  const int mt = blockIdx.x * 2 + (w >> 2);
  const int nt = w & 3;
  const int row0 = mt * 16, n0 = nt * 16;

  const int b = row0 >> 9;            // row = b*T + t; b uniform over tile
  const int t = (row0 & 511) + lo;    // this lane's A-row time index

  const float bias = fcb[n0 + lo];
  v8f acc;
#pragma unroll
  for (int v = 0; v < 8; ++v) acc[v] = bias;

  const __bf16* brow = fcw_bf + (size_t)(n0 + lo) * (2 * Hh);

#pragma unroll
  for (int kb = 0; kb < 8; ++kb) {
    const int dirk = kb >> 2;       // k<128 -> forward h, k>=128 -> backward h
    const int kl = (kb & 3) * 32;
    const __bf16* arow = hs_all + (size_t)dirk * Tt * Bb * Hh +
                         (size_t)t * Bb * Hh + b * Hh + kl;
    v16bf a = make_a16(arow + hi * 8, arow + 16 + hi * 8);
    v16bf bm = make_a16(brow + kb * 32 + hi * 8, brow + kb * 32 + 16 + hi * 8);
    acc = __builtin_amdgcn_wmma_f32_16x16x32_bf16(false, a, false, bm,
                                                  (short)0, acc, false, false);
  }
#pragma unroll
  for (int v = 0; v < 8; ++v) {
    const int tt = (row0 & 511) + hi * 8 + v;
    out[((size_t)b * Tt + tt) * Vv + n0 + lo] = acc[v];
  }
}

// ---------------------------------------------------------------------------
extern "C" void kernel_launch(void* const* d_in, const int* in_sizes, int n_in,
                              void* d_out, int out_size, void* d_ws, size_t ws_size,
                              hipStream_t stream) {
  const int*   x    = (const int*)d_in[0];
  const float* emb  = (const float*)d_in[1];
  const float* wihf = (const float*)d_in[2];
  const float* whhf = (const float*)d_in[3];
  const float* bihf = (const float*)d_in[4];
  const float* bhhf = (const float*)d_in[5];
  const float* wihb = (const float*)d_in[6];
  const float* whhb = (const float*)d_in[7];
  const float* bihb = (const float*)d_in[8];
  const float* bhhb = (const float*)d_in[9];
  const float* fcw  = (const float*)d_in[10];
  const float* fcb  = (const float*)d_in[11];
  float* out = (float*)d_out;

  char* ws = (char*)d_ws;
  size_t o = 0;
  __bf16* xp    = (__bf16*)(ws + o); o += (size_t)2 * Tt * Bb * G4H * 2;  // 134.2 MB
  __bf16* hs    = (__bf16*)(ws + o); o += (size_t)2 * Tt * Bb * Hh * 2;   //  33.6 MB
  __bf16* embbf = (__bf16*)(ws + o); o += (size_t)Vv * Hh * 2;
  __bf16* wihbf = (__bf16*)(ws + o); o += (size_t)2 * G4H * Hh * 2;
  __bf16* whhbf = (__bf16*)(ws + o); o += (size_t)2 * G4H * Hh * 2;
  __bf16* fcwbf = (__bf16*)(ws + o); o += (size_t)Vv * 2 * Hh * 2;
  float*  bsum  = (float*)(ws + o);  o += (size_t)2 * G4H * 4;

  prep_kernel<<<256, 256, 0, stream>>>(emb, wihf, whhf, bihf, bhhf,
                                       wihb, whhb, bihb, bhhb, fcw,
                                       embbf, wihbf, whhbf, fcwbf, bsum);

  proj_kernel<<<dim3(4096, 4, 2), 256, 0, stream>>>(x, embbf, wihbf, bsum, xp);

  (void)hipFuncSetAttribute((const void*)lstm_kernel,
                            hipFuncAttributeMaxDynamicSharedMemorySize, 196608);
  lstm_kernel<<<2, 1024, 196608, stream>>>(xp, whhbf, hs);

  fc_kernel<<<2048, 256, 0, stream>>>(hs, fcwbf, fcb, out);
}
